// TransformerBlock_541165879394
// MI455X (gfx1250) — matrix operand order repaired and checked by reading, not compiled, not run on hardware
//
#include <hip/hip_runtime.h>
#include <hip/hip_bf16.h>
#include <stdint.h>

// ---------------------------------------------------------------------------
// Types for CDNA5 WMMA
// ---------------------------------------------------------------------------
typedef __attribute__((ext_vector_type(16))) __bf16 bf16x16;
typedef __attribute__((ext_vector_type(8)))  float  f32x8;

union FragU { uint4 u[2]; bf16x16 v; };

static __device__ __forceinline__ __bf16 f2bf(float f) {
    union { float f; uint32_t u; } in; in.f = f;
    uint32_t r = in.u + 0x7FFFu + ((in.u >> 16) & 1u);   // round-to-nearest-even
    unsigned short h = (unsigned short)(r >> 16);
    union { unsigned short s; __bf16 b; } out; out.s = h;
    return out.b;
}

// Generic->LDS address: low 32 bits of a generic LDS pointer are the
// group-segment offset (ISA 10.2).
static __device__ __forceinline__ uint32_t lds_off(const void* p) {
    return (uint32_t)(uintptr_t)p;
}

// CDNA5 async global->LDS DMA (ASYNCcnt-tracked), GVS addressing:
// mem = SADDR64 + VADDR32 ; LDS[vdst_lane] <- 16 bytes.
static __device__ __forceinline__ void async_ld_b128(uint32_t ldsoff, const void* saddr, uint32_t voff) {
    asm volatile("global_load_async_to_lds_b128 %0, %1, %2"
                 :: "v"(ldsoff), "v"(voff), "s"(saddr) : "memory");
}
static __device__ __forceinline__ void wait_async0() {
    asm volatile("s_wait_asynccnt 0" ::: "memory");
}

// Problem constants (B=1)
#define TT   2048
#define DD   1024
#define HH   16
#define DH   64
#define DFF  4096
#define NEG_BIG (-3.0e38f)
#define LDSW 40   // padded LDS row stride (32 data + 8 pad bf16) to spread banks

// ---------------------------------------------------------------------------
// Weight conversion kernels (f32 -> bf16, transposed to N x K row-major)
// ---------------------------------------------------------------------------
__global__ void conv_qkvw(const float* __restrict__ wq, const float* __restrict__ wk,
                          const float* __restrict__ wv, __bf16* __restrict__ WqkvT) {
    int idx = blockIdx.x * 256 + threadIdx.x;   // 3072*1024 total
    int n = idx >> 10, d = idx & 1023;
    int sel = n >> 10, nn = n & 1023;
    int h = nn >> 6, e = nn & 63;
    const float* w = (sel == 0) ? wq : ((sel == 1) ? wk : wv);
    WqkvT[idx] = f2bf(w[h * (DD * DH) + d * DH + e]);
}

__global__ void conv_T(const float* __restrict__ in, __bf16* __restrict__ out, int N, int K) {
    size_t idx = (size_t)blockIdx.x * 256 + threadIdx.x;  // N*K total
    int n = (int)(idx / K), kk = (int)(idx % K);
    out[idx] = f2bf(in[(size_t)kk * N + n]);
}

// ---------------------------------------------------------------------------
// LayerNorm: torch-style (std with ddof=1, eps added to std) -> bf16
// ---------------------------------------------------------------------------
__global__ __launch_bounds__(256) void ln_kernel(const float* __restrict__ x,
                                                 const float* __restrict__ g,
                                                 const float* __restrict__ b,
                                                 __bf16* __restrict__ out) {
    int row = blockIdx.x, tid = threadIdx.x;
    const float* xr = x + (size_t)row * DD;
    float v[4];
#pragma unroll
    for (int i = 0; i < 4; i++) v[i] = xr[tid + i * 256];
    __shared__ float red[256];
    float s = v[0] + v[1] + v[2] + v[3];
    red[tid] = s; __syncthreads();
    for (int st = 128; st > 0; st >>= 1) { if (tid < st) red[tid] += red[tid + st]; __syncthreads(); }
    float mean = red[0] * (1.0f / DD);
    __syncthreads();
    float ss = 0.f;
#pragma unroll
    for (int i = 0; i < 4; i++) { float d = v[i] - mean; ss += d * d; }
    red[tid] = ss; __syncthreads();
    for (int st = 128; st > 0; st >>= 1) { if (tid < st) red[tid] += red[tid + st]; __syncthreads(); }
    float stdv = sqrtf(red[0] / (float)(DD - 1));
    float inv = 1.0f / (stdv + 1e-8f);
#pragma unroll
    for (int i = 0; i < 4; i++) {
        int c = tid + i * 256;
        out[(size_t)row * DD + c] = f2bf((v[i] - mean) * inv * g[c] + b[c]);
    }
}

// ---------------------------------------------------------------------------
// Tiled bf16 WMMA GEMM:  C[M,N] = A[M,K] * BT[N,K]^T   (f32 accumulate)
// 256 threads = 8 waves; wave tile 64x32 (8 accs, 8 WMMA/K-step);
// block tile 128x128; K-step 32; async global->LDS DMA, double-buffered,
// one barrier per K-step.
// MODE 0: QKV scatter (q,k row-major per head; v stored transposed Dh x T)
// MODE 1: out = acc + bias[col] + resid   (f32 store)
// MODE 2: out = gelu_exact(acc + bias[col]) -> bf16
// ---------------------------------------------------------------------------
template <int MODE>
__global__ __launch_bounds__(256) void gemm_bf16(
    const __bf16* __restrict__ A, const __bf16* __restrict__ BT,
    int M, int N, int K,
    const float* __restrict__ bias, const float* __restrict__ resid,
    float* __restrict__ outF, __bf16* __restrict__ outBf,
    __bf16* __restrict__ outQ, __bf16* __restrict__ outK, __bf16* __restrict__ outVT) {
    __shared__ alignas(16) __bf16 sA[2][128 * LDSW];
    __shared__ alignas(16) __bf16 sB[2][128 * LDSW];
    int tid = threadIdx.x;
    int wave = tid >> 5, lane = tid & 31, l = lane & 15, hi = lane >> 4;
    int wm = wave & 1, wn = wave >> 1;           // 2 x 4 wave grid
    int m0 = blockIdx.y * 128, n0 = blockIdx.x * 128;

    uint32_t aBase[2] = { lds_off(&sA[0][0]), lds_off(&sA[1][0]) };
    uint32_t bBase[2] = { lds_off(&sB[0][0]), lds_off(&sB[1][0]) };

    // staging pattern: 512 chunks of 8 bf16 per tile, 2 chunks/thread/tile
    int srow0 = tid >> 2, sch0 = (tid & 3) * 8;
    int srow1 = (tid + 256) >> 2, sch1 = ((tid + 256) & 3) * 8;

#define STAGE(buf, kk)                                                                        \
    do {                                                                                      \
        async_ld_b128(aBase[buf] + (uint32_t)(srow0 * LDSW + sch0) * 2, A,                    \
                      (uint32_t)((m0 + srow0) * K + (kk) + sch0) * 2);                        \
        async_ld_b128(aBase[buf] + (uint32_t)(srow1 * LDSW + sch1) * 2, A,                    \
                      (uint32_t)((m0 + srow1) * K + (kk) + sch1) * 2);                        \
        async_ld_b128(bBase[buf] + (uint32_t)(srow0 * LDSW + sch0) * 2, BT,                   \
                      (uint32_t)((n0 + srow0) * K + (kk) + sch0) * 2);                        \
        async_ld_b128(bBase[buf] + (uint32_t)(srow1 * LDSW + sch1) * 2, BT,                   \
                      (uint32_t)((n0 + srow1) * K + (kk) + sch1) * 2);                        \
    } while (0)

    f32x8 acc[4][2] = {};

    STAGE(0, 0);
    wait_async0();
    __syncthreads();

    for (int kk = 0; kk < K; kk += 32) {
        int buf = (kk >> 5) & 1;
        if (kk + 32 < K) STAGE(buf ^ 1, kk + 32);

        bf16x16 af[4], bg[2];
#pragma unroll
        for (int f = 0; f < 4; f++) {   // A frag: lane<16 K{0..7,16..23}; hi K{8..15,24..31}
            const __bf16* base = &sA[buf][(wm * 64 + f * 16 + l) * LDSW];
            FragU u;
            u.u[0] = *(const uint4*)(base + 8 * hi);
            u.u[1] = *(const uint4*)(base + 16 + 8 * hi);
            af[f] = u.v;
        }
#pragma unroll
        for (int g = 0; g < 2; g++) {   // B frag: lane<16 K{0..7,16..23}; hi K{8..15,24..31}
            const __bf16* base = &sB[buf][(wn * 32 + g * 16 + l) * LDSW];
            FragU u;
            u.u[0] = *(const uint4*)(base + 8 * hi);
            u.u[1] = *(const uint4*)(base + 16 + 8 * hi);
            bg[g] = u.v;
        }
#pragma unroll
        for (int f = 0; f < 4; f++)
#pragma unroll
            for (int g = 0; g < 2; g++)
                acc[f][g] = __builtin_amdgcn_wmma_f32_16x16x32_bf16(
                    false, af[f], false, bg[g], (short)0, acc[f][g], false, false);

        wait_async0();       // own async loads into next buffer complete
        __syncthreads();     // all waves done reading cur + done loading next
    }
#undef STAGE

#pragma unroll
    for (int f = 0; f < 4; f++) {
#pragma unroll
        for (int g = 0; g < 2; g++) {
            int col = n0 + wn * 32 + g * 16 + l;
#pragma unroll
            for (int r = 0; r < 8; r++) {
                int row = m0 + wm * 64 + f * 16 + r + 8 * hi;
                float val = acc[f][g][r];
                if (MODE == 0) {
                    int sel = col >> 10, nn = col & 1023, h = nn >> 6, e = nn & 63;
                    __bf16 bv = f2bf(val);
                    if (sel == 0)      outQ [(size_t)h * TT * DH + (size_t)row * DH + e] = bv;
                    else if (sel == 1) outK [(size_t)h * TT * DH + (size_t)row * DH + e] = bv;
                    else               outVT[(size_t)h * DH * TT + (size_t)e * TT + row] = bv;
                } else if (MODE == 1) {
                    outF[(size_t)row * N + col] = val + bias[col] + resid[(size_t)row * N + col];
                } else {
                    float t = val + bias[col];
                    float gel = 0.5f * t * (1.0f + erff(t * 0.70710678118654752f));
                    outBf[(size_t)row * N + col] = f2bf(gel);
                }
            }
        }
    }
}

// ---------------------------------------------------------------------------
// Flash attention (causal). One wave per (head, 16-query tile).
// Computes S^T = K*Q^T so softmax reduction is in-lane; ctx^T = V^T * P^T.
// All fragments loaded directly from global (b128); no LDS, no block sync.
// ---------------------------------------------------------------------------
__global__ __launch_bounds__(256) void attn_kernel(
    const __bf16* __restrict__ q, const __bf16* __restrict__ k,
    const __bf16* __restrict__ vT, __bf16* __restrict__ ctx) {
    int wave = threadIdx.x >> 5, lane = threadIdx.x & 31;
    int l = lane & 15, hi = lane >> 4;
    int task = blockIdx.x * 8 + wave;           // H * (T/16) = 2048 tasks
    int head = task >> 7;
    int t0 = (task & 127) << 4;

    const __bf16* qh = q  + (size_t)head * TT * DH;
    const __bf16* kh = k  + (size_t)head * TT * DH;
    const __bf16* vh = vT + (size_t)head * DH * TT;

    // Q^T B-fragments (2 e-steps of 32): lane<16 e=base+{0..7,16..23}, hi e=base+{8..15,24..31}
    bf16x16 bQ[2];
#pragma unroll
    for (int es = 0; es < 2; es++) {
        size_t base = (size_t)(t0 + l) * DH + es * 32 + 8 * hi;
        FragU u;
        u.u[0] = *(const uint4*)&qh[base];
        u.u[1] = *(const uint4*)&qh[base + 16];
        bQ[es] = u.v;
    }

    float m = NEG_BIG, suml = 0.f;
    f32x8 accC[4] = {};                          // ctx^T, 4 e-blocks of 16
    int tq = t0 + l;
    int nsteps = (t0 + 47) >> 5;                 // keys 0 .. t0+15, step 32

    for (int st = 0; st < nsteps; st++) {
        int s0 = st << 5;
        if (st + 1 < nsteps) {                   // prefetch next key-step K/V rows
            int sn = s0 + 32;
            __builtin_prefetch(&kh[(size_t)(sn + lane) * DH], 0, 0);
            __builtin_prefetch(&vh[(size_t)(lane * 2) * TT + sn], 0, 0);
        }
        // ---- scores: S^T tiles (keys s0..s0+15 and s0+16..s0+31) ----
        f32x8 sc0 = {}, sc1 = {};
#pragma unroll
        for (int es = 0; es < 2; es++) {
            FragU a0, a1;
            size_t b0 = (size_t)(s0 + l) * DH + es * 32 + 8 * hi;
            size_t b1 = (size_t)(s0 + 16 + l) * DH + es * 32 + 8 * hi;
            a0.u[0] = *(const uint4*)&kh[b0];      a0.u[1] = *(const uint4*)&kh[b0 + 16];
            a1.u[0] = *(const uint4*)&kh[b1];      a1.u[1] = *(const uint4*)&kh[b1 + 16];
            sc0 = __builtin_amdgcn_wmma_f32_16x16x32_bf16(false, a0.v, false, bQ[es], (short)0, sc0, false, false);
            sc1 = __builtin_amdgcn_wmma_f32_16x16x32_bf16(false, a1.v, false, bQ[es], (short)0, sc1, false, false);
        }
        // ---- causal mask + scale + online softmax ----
        float p0[8], p1[8], mloc = NEG_BIG;
#pragma unroll
        for (int r = 0; r < 8; r++) {
            int sk0 = s0 + r + 8 * hi;
            int sk1 = s0 + 16 + r + 8 * hi;
            float v0 = sc0[r] * 0.125f; if (sk0 > tq) v0 = NEG_BIG;
            float v1 = sc1[r] * 0.125f; if (sk1 > tq) v1 = NEG_BIG;
            p0[r] = v0; p1[r] = v1;
            mloc = fmaxf(mloc, fmaxf(v0, v1));
        }
        mloc = fmaxf(mloc, __shfl_xor(mloc, 16, 32));
        float mnew = fmaxf(m, mloc);
        float corr = __expf(m - mnew);
        m = mnew;
        float ls = 0.f;
#pragma unroll
        for (int r = 0; r < 8; r++) {
            p0[r] = __expf(p0[r] - mnew);
            p1[r] = __expf(p1[r] - mnew);
            ls += p0[r] + p1[r];
        }
        ls += __shfl_xor(ls, 16, 32);
        suml = suml * corr + ls;
#pragma unroll
        for (int eb = 0; eb < 4; eb++)
#pragma unroll
            for (int r = 0; r < 8; r++) accC[eb][r] *= corr;
        // ---- build P^T B-fragment (32 keys x 16 queries) via lane-pair exchange ----
        bf16x16 pB;
#pragma unroll
        for (int r = 0; r < 8; r++) {
            float o0 = __shfl_xor(p0[r], 16, 32);
            float o1 = __shfl_xor(p1[r], 16, 32);
            if (hi == 0) { pB[r] = f2bf(p0[r]); pB[8 + r] = f2bf(o0); }   // keys s0..s0+15
            else         { pB[r] = f2bf(o1);    pB[8 + r] = f2bf(p1[r]); } // keys s0+16..s0+31
        }
        // ---- ctx^T += V^T(e,keys) * P^T ----
#pragma unroll
        for (int eb = 0; eb < 4; eb++) {
            FragU a;
            size_t base = (size_t)(eb * 16 + l) * TT + s0 + 16 * hi;
            a.u[0] = *(const uint4*)&vh[base];
            a.u[1] = *(const uint4*)&vh[base + 8];
            accC[eb] = __builtin_amdgcn_wmma_f32_16x16x32_bf16(false, a.v, false, pB, (short)0, accC[eb], false, false);
        }
    }
    // ---- finalize: ctx[t][head*64 + e] ----
    float inv = 1.0f / suml;
#pragma unroll
    for (int eb = 0; eb < 4; eb++)
#pragma unroll
        for (int r = 0; r < 8; r++) {
            int e = eb * 16 + r + 8 * hi;
            ctx[(size_t)(t0 + l) * DD + head * DH + e] = f2bf(accC[eb][r] * inv);
        }
}

// ---------------------------------------------------------------------------
// Host launch
// ---------------------------------------------------------------------------
extern "C" void kernel_launch(void* const* d_in, const int* in_sizes, int n_in,
                              void* d_out, int out_size, void* d_ws, size_t ws_size,
                              hipStream_t stream) {
    const float* x     = (const float*)d_in[0];
    const float* wq    = (const float*)d_in[1];
    const float* wk    = (const float*)d_in[2];
    const float* wv    = (const float*)d_in[3];
    const float* wo    = (const float*)d_in[4];
    const float* bo    = (const float*)d_in[5];
    const float* ln1_g = (const float*)d_in[6];
    const float* ln1_b = (const float*)d_in[7];
    const float* ln2_g = (const float*)d_in[8];
    const float* ln2_b = (const float*)d_in[9];
    const float* w1    = (const float*)d_in[10];
    const float* b1    = (const float*)d_in[11];
    const float* w2    = (const float*)d_in[12];
    const float* b2    = (const float*)d_in[13];
    float* out = (float*)d_out;

    char* w = (char*)d_ws;
    size_t off = 0;
    auto nxt = [&](size_t bytes) { void* p = w + off; off += (bytes + 255) & ~(size_t)255; return p; };

    __bf16* WqkvT = (__bf16*)nxt((size_t)3 * DD * DD * 2);    // 3072 x 1024
    __bf16* WoT   = (__bf16*)nxt((size_t)DD * DD * 2);        // 1024 x 1024
    __bf16* W1T   = (__bf16*)nxt((size_t)DFF * DD * 2);       // 4096 x 1024
    __bf16* W2T   = (__bf16*)nxt((size_t)DD * DFF * 2);       // 1024 x 4096
    __bf16* h1    = (__bf16*)nxt((size_t)TT * DD * 2);
    __bf16* qb    = (__bf16*)nxt((size_t)HH * TT * DH * 2);
    __bf16* kb    = (__bf16*)nxt((size_t)HH * TT * DH * 2);
    __bf16* vTb   = (__bf16*)nxt((size_t)HH * DH * TT * 2);
    __bf16* ctxb  = (__bf16*)nxt((size_t)TT * DD * 2);
    float*  out1  = (float*)nxt((size_t)TT * DD * 4);
    __bf16* h2    = (__bf16*)nxt((size_t)TT * DD * 2);
    __bf16* actb  = (__bf16*)nxt((size_t)TT * DFF * 2);

    // 1) weight conversion / transposition
    conv_qkvw<<<(3 * DD * DD) / 256, 256, 0, stream>>>(wq, wk, wv, WqkvT);
    conv_T<<<(DD * DD) / 256, 256, 0, stream>>>(wo, WoT, DD, DD);
    conv_T<<<(DFF * DD) / 256, 256, 0, stream>>>(w1, W1T, DFF, DD);
    conv_T<<<(DD * DFF) / 256, 256, 0, stream>>>(w2, W2T, DD, DFF);

    // 2) LN1
    ln_kernel<<<TT, 256, 0, stream>>>(x, ln1_g, ln1_b, h1);

    // 3) QKV GEMM (M=2048, N=3072, K=1024) with scatter epilogue
    gemm_bf16<0><<<dim3(3 * DD / 128, TT / 128), 256, 0, stream>>>(
        h1, WqkvT, TT, 3 * DD, DD, nullptr, nullptr, nullptr, nullptr, qb, kb, vTb);

    // 4) causal flash attention
    attn_kernel<<<(HH * (TT / 16)) / 8, 256, 0, stream>>>(qb, kb, vTb, ctxb);

    // 5) output projection + residual: out1 = x + ctx @ wo + bo
    gemm_bf16<1><<<dim3(DD / 128, TT / 128), 256, 0, stream>>>(
        ctxb, WoT, TT, DD, DD, bo, x, out1, nullptr, nullptr, nullptr, nullptr);

    // 6) LN2
    ln_kernel<<<TT, 256, 0, stream>>>(out1, ln2_g, ln2_b, h2);

    // 7) FFN1 + exact GELU -> bf16
    gemm_bf16<2><<<dim3(DFF / 128, TT / 128), 256, 0, stream>>>(
        h2, W1T, TT, DFF, DD, b1, nullptr, nullptr, actb, nullptr, nullptr, nullptr);

    // 8) FFN2 + residual: out = out1 + act @ w2 + b2
    gemm_bf16<1><<<dim3(DD / 128, TT / 128), 256, 0, stream>>>(
        actb, W2T, TT, DD, DFF, b2, out1, out, nullptr, nullptr, nullptr, nullptr);

    (void)in_sizes; (void)n_in; (void)out_size; (void)ws_size;
}
